// Network_49855980372366
// MI455X (gfx1250) — matrix operand order repaired and checked by reading, not compiled, not run on hardware
//
#include <hip/hip_runtime.h>
#include <math.h>

typedef _Float16 h16;
typedef __attribute__((ext_vector_type(16))) _Float16 v16h;
typedef __attribute__((ext_vector_type(8)))  _Float16 v8h;
typedef __attribute__((ext_vector_type(8)))  float    v8f;

#define BATCH 8

// Packed B-fragment addressing: X stored as [z][kTile][pTile][lane][16 halfs].
// Within a 32-K chunk: K offset k2 -> lane group (k2>>4), element (k2&15).
static __device__ __forceinline__ long pkh(int z, int c, long pos, int nKt, long nPt) {
  int kt = c >> 5, k2 = c & 31;
  int half = k2 >> 4, e = k2 & 15;
  long pt = pos >> 4; int col = (int)(pos & 15);
  return ((((long)z * nKt + kt) * nPt + pt) * 32 + col + 16 * half) * 16 + e;
}

// ---------------------------------------------------------------------------
// Pack A (f32, arbitrary (o,k) strides, optional batch) into WMMA A-fragment
// layout [z][oTile][kTile][lane][16 halfs], zero-padded to K multiple of 32.
// ---------------------------------------------------------------------------
__global__ void pack_A(const float* __restrict__ A, long sAo, long sAk, long sAb,
                       h16* __restrict__ out, int O, int K, int Z) {
  int nKt = (K + 31) >> 5;
  int nOt = O >> 4;
  long total = (long)Z * nOt * nKt * 32;
  long t = (long)blockIdx.x * blockDim.x + threadIdx.x;
  if (t >= total) return;
  int lane = (int)(t & 31);
  long q = t >> 5;
  int kt = (int)(q % nKt); q /= nKt;
  int ot = (int)(q % nOt);
  int z  = (int)(q / nOt);
  int row = lane & 15, hl = lane >> 4;
  v16h f;
#pragma unroll
  for (int e = 0; e < 16; ++e) {
    int kk = e + 16 * hl;
    int k = kt * 32 + kk;
    f[e] = (k < K) ? (h16)A[(long)z * sAb + (long)(ot * 16 + row) * sAo + (long)k * sAk]
                   : (h16)0.f;
  }
  ((v16h*)out)[t] = f;
}

// Pack X (planar f32, row stride sXk, batched) into B-fragment layout, K zero-padded.
__global__ void pack_B(const float* __restrict__ X, long sXk, long sXb,
                       h16* __restrict__ out, int K, int P, int Z) {
  int nKt = (K + 31) >> 5;
  long nPt = P >> 4;
  long total = (long)Z * nKt * nPt * 32;
  long t = (long)blockIdx.x * blockDim.x + threadIdx.x;
  if (t >= total) return;
  int lane = (int)(t & 31);
  long q = t >> 5;
  long pt = q % nPt; q /= nPt;
  int kt = (int)(q % nKt);
  int z  = (int)(q / nKt);
  int col = lane & 15, half = lane >> 4;
  v16h f;
#pragma unroll
  for (int e = 0; e < 16; ++e) {
    int k = kt * 32 + e + 16 * half;
    f[e] = (k < K) ? (h16)X[(long)z * sXb + (long)k * sXk + pt * 16 + col] : (h16)0.f;
  }
  ((v16h*)out)[t] = f;
}

// ---------------------------------------------------------------------------
// Batched WMMA GEMM, both operands pre-packed:
//   OUT[o,p] = epi( sum_k A[o,k] * X[k,p] )
// One wave computes a 16(O) x 64(P) strip: 4 accumulators sharing one A frag.
// A tile: one 32B load/lane. B tile: one 32B load/lane. No guards in the loop.
// FULL: P % 64 == 0 -> straight-line inner loop (no exec-mask branching).
// PACKED_OUT: write D as f16 in the B-fragment layout of the next GEMM
// (requires O multiple of 32); each lane stores its 8 values as one 16B store.
// ---------------------------------------------------------------------------
template<typename TOUT, bool PACKED_OUT, bool FULL,
         bool HAS_B1, bool HAS_SB, bool RELU, bool HAS_RES>
__global__ void __launch_bounds__(128) wmma_gemm(
    const h16* __restrict__ Apk, long sApb,          // packed A batch stride (v16h units)
    const h16* __restrict__ Xpk, long sXpb,          // packed X batch stride (v16h units)
    TOUT* __restrict__ OUT, long sOo, long sOb,      // planar strides (unused if PACKED_OUT)
    const float* __restrict__ b1,
    const float* __restrict__ scale, float scaleMul,
    const float* __restrict__ b2,
    const float* __restrict__ RES, long sRo, long sRb,
    int O, int K, int P)
{
  int wave = threadIdx.x >> 5;
  int lane = threadIdx.x & 31;
  int p0 = (blockIdx.x * 4 + wave) * 64;
  if (p0 >= P) return;
  int col = lane & 15, hl = lane >> 4;
  int nKt = (K + 31) >> 5;
  long nPt = P >> 4;
  long pt0 = p0 >> 4;
  const v16h* Ap = (const v16h*)Apk + (long)blockIdx.z * sApb
                   + (long)blockIdx.y * nKt * 32 + lane;
  const v16h* Xp = (const v16h*)Xpk + (long)blockIdx.z * sXpb + lane;

  bool act[4];
#pragma unroll
  for (int t = 0; t < 4; ++t) act[t] = FULL ? true : ((p0 + t * 16) < P);

  v8f zero = {0.f, 0.f, 0.f, 0.f, 0.f, 0.f, 0.f, 0.f};
  v8f acc[4] = {zero, zero, zero, zero};

  for (int kt = 0; kt < nKt; ++kt) {
    v16h a = Ap[(long)kt * 32];
    if (FULL) {
      const v16h* Xt = Xp + ((long)kt * nPt + pt0) * 32;
      v16h b0 = Xt[0];
      v16h b1v = Xt[32];
      v16h b2v = Xt[64];
      v16h b3v = Xt[96];
      if (kt + 1 < nKt)
        __builtin_prefetch((const void*)(Ap + (long)(kt + 1) * 32), 0, 3);
      acc[0] = __builtin_amdgcn_wmma_f32_16x16x32_f16(false, a, false, b0, (short)0,
                                                      acc[0], false, false);
      acc[1] = __builtin_amdgcn_wmma_f32_16x16x32_f16(false, a, false, b1v, (short)0,
                                                      acc[1], false, false);
      acc[2] = __builtin_amdgcn_wmma_f32_16x16x32_f16(false, a, false, b2v, (short)0,
                                                      acc[2], false, false);
      acc[3] = __builtin_amdgcn_wmma_f32_16x16x32_f16(false, a, false, b3v, (short)0,
                                                      acc[3], false, false);
    } else {
#pragma unroll
      for (int t = 0; t < 4; ++t) {
        if (!act[t]) continue;
        v16h b = Xp[((long)kt * nPt + pt0 + t) * 32];
        acc[t] = __builtin_amdgcn_wmma_f32_16x16x32_f16(false, a, false, b, (short)0,
                                                        acc[t], false, false);
      }
    }
  }

  // epilogue constants per output row
  int o0 = blockIdx.y * 16;
  float vb1[8], vsc[8], vb2[8];
#pragma unroll
  for (int r = 0; r < 8; ++r) {
    int o = o0 + r + 8 * hl;
    vb1[r] = HAS_B1 ? b1[o] : 0.f;
    vsc[r] = HAS_SB ? scale[o] * scaleMul : 1.f;
    vb2[r] = HAS_SB ? b2[o] : 0.f;
  }

  if (PACKED_OUT) {
    v8h* Op = (v8h*)OUT;
    int lane2 = col + 16 * (blockIdx.y & 1);   // o-tile parity selects lane group
    long ktp = blockIdx.y >> 1;                // K' chunk of next GEMM
#pragma unroll
    for (int t = 0; t < 4; ++t) {
      if (!act[t]) continue;
      v8h pk;
#pragma unroll
      for (int r = 0; r < 8; ++r) {
        float v = acc[t][r];
        if (HAS_B1) v += vb1[r];
        if (HAS_SB) v = v * vsc[r] + vb2[r];
        if (RELU)   v = fmaxf(v, 0.f);
        pk[r] = (h16)v;
      }
      Op[((ktp * nPt + pt0 + t) * 32 + lane2) * 2 + hl] = pk;
    }
  } else {
    const float* Rb = HAS_RES ? (RES + (long)blockIdx.z * sRb) : nullptr;
    TOUT* Ob = OUT + (long)blockIdx.z * sOb;
#pragma unroll
    for (int t = 0; t < 4; ++t) {
      if (!act[t]) continue;
#pragma unroll
      for (int r = 0; r < 8; ++r) {
        int o = o0 + r + 8 * hl;
        float v = acc[t][r];
        if (HAS_B1) v += vb1[r];
        if (HAS_SB) v = v * vsc[r] + vb2[r];
        if (RELU)   v = fmaxf(v, 0.f);
        if (HAS_RES) v += Rb[(long)o * sRo + p0 + t * 16 + col];
        Ob[(long)o * sOo + p0 + t * 16 + col] = (TOUT)v;
      }
    }
  }
}

// ---------------------------------------------------------------------------
// pc (B,N,3) -> feat0 (B,3,N)
__global__ void transpose_pc(const float* __restrict__ x, float* __restrict__ f0, int N) {
  int t = blockIdx.x * blockDim.x + threadIdx.x;
  if (t >= BATCH * N) return;
  int b = t / N, n = t % N;
  for (int c = 0; c < 3; ++c)
    f0[((long)b * 3 + c) * N + n] = x[((long)b * N + n) * 3 + c];
}

// Furthest point sampling: one block per batch, LDS-resident min-distance array.
__global__ void fps_kernel(const float* __restrict__ xyz, int* __restrict__ out, int N, int M) {
  __shared__ float sd[4096];
  __shared__ float rv[256];
  __shared__ int   ri[256];
  int b = blockIdx.x;
  const float* P = xyz + (long)b * N * 3;
  for (int j = threadIdx.x; j < N; j += blockDim.x) sd[j] = 1e10f;
  if (threadIdx.x == 0) out[b * M] = 0;
  __syncthreads();
  int last = 0;
  for (int s = 1; s < M; ++s) {
    float lx = P[last * 3 + 0], ly = P[last * 3 + 1], lz = P[last * 3 + 2];
    float bm = -1.f; int bi = 0x7fffffff;
    for (int j = threadIdx.x; j < N; j += blockDim.x) {
      float dx = P[j * 3 + 0] - lx, dy = P[j * 3 + 1] - ly, dz = P[j * 3 + 2] - lz;
      float d = dx * dx + dy * dy + dz * dz;
      float mn = fminf(sd[j], d);
      sd[j] = mn;
      if (mn > bm) { bm = mn; bi = j; }
    }
    rv[threadIdx.x] = bm; ri[threadIdx.x] = bi;
    __syncthreads();
    for (int off = 128; off > 0; off >>= 1) {
      if (threadIdx.x < off) {
        float ov = rv[threadIdx.x + off]; int oi = ri[threadIdx.x + off];
        if (ov > rv[threadIdx.x] || (ov == rv[threadIdx.x] && oi < ri[threadIdx.x])) {
          rv[threadIdx.x] = ov; ri[threadIdx.x] = oi;
        }
      }
      __syncthreads();
    }
    last = ri[0];
    if (threadIdx.x == 0) out[b * M + s] = last;
    __syncthreads();
  }
}

__global__ void gather_xyz(const float* __restrict__ xyz, const int* __restrict__ cidx,
                           float* __restrict__ nxyz, int N, int M) {
  int t = blockIdx.x * blockDim.x + threadIdx.x;
  if (t >= BATCH * M) return;
  int b = t / M;
  int j = cidx[t];
  for (int c = 0; c < 3; ++c)
    nxyz[(long)t * 3 + c] = xyz[((long)b * N + j) * 3 + c];
}

// kNN: one block per query; distances in LDS, 32 rounds of argmin extraction.
__global__ void knn_kernel(const float* __restrict__ xyz, const float* __restrict__ nx,
                           int* __restrict__ knn, int N, int M) {
  __shared__ float sd[4096];
  __shared__ float rv[128];
  __shared__ int   ri[128];
  int q = blockIdx.x;
  int b = q / M;
  const float* P = xyz + (long)b * N * 3;
  float qx = nx[(long)q * 3 + 0], qy = nx[(long)q * 3 + 1], qz = nx[(long)q * 3 + 2];
  for (int j = threadIdx.x; j < N; j += blockDim.x) {
    float dx = P[j * 3 + 0] - qx, dy = P[j * 3 + 1] - qy, dz = P[j * 3 + 2] - qz;
    sd[j] = dx * dx + dy * dy + dz * dz;
  }
  __syncthreads();
  for (int kk = 0; kk < 32; ++kk) {
    float bv = 3.0e38f; int bi = 0x7fffffff;
    for (int j = threadIdx.x; j < N; j += blockDim.x) {
      float v = sd[j];
      if (v < bv || (v == bv && j < bi)) { bv = v; bi = j; }
    }
    rv[threadIdx.x] = bv; ri[threadIdx.x] = bi;
    __syncthreads();
    for (int off = 64; off > 0; off >>= 1) {
      if (threadIdx.x < off) {
        float ov = rv[threadIdx.x + off]; int oi = ri[threadIdx.x + off];
        if (ov < rv[threadIdx.x] || (ov == rv[threadIdx.x] && oi < ri[threadIdx.x])) {
          rv[threadIdx.x] = ov; ri[threadIdx.x] = oi;
        }
      }
      __syncthreads();
    }
    if (threadIdx.x == 0) { knn[(long)q * 32 + kk] = ri[0]; sd[ri[0]] = 3.4e38f; }
    __syncthreads();
  }
}

// nf = concat(rel_xyz, feats - center feats) written directly in packed B layout,
// zero-padded to Kpad channels.
__global__ void edge_feat(const float* __restrict__ xyz, const float* __restrict__ feats,
                          const float* __restrict__ nxyz, const int* __restrict__ cidx,
                          const int* __restrict__ knn, h16* __restrict__ nf,
                          int Np, int M, int Cpc, int Kpad, int Ptot) {
  int t = blockIdx.x * blockDim.x + threadIdx.x;
  if (t >= Ptot) return;
  long pos = t;
  int nKt = Kpad >> 5;
  long nPt = Ptot >> 4;
  int kk = t & 31;
  int bm = t >> 5;            // b*M + m
  int b = bm / M;
  int j = knn[(long)bm * 32 + kk];
  int cj = cidx[bm];
  const float* X  = xyz + (long)b * Np * 3;
  const float* NX = nxyz + (long)bm * 3;
  nf[pkh(0, 0, pos, nKt, nPt)] = (h16)(X[j * 3 + 0] - NX[0]);
  nf[pkh(0, 1, pos, nKt, nPt)] = (h16)(X[j * 3 + 1] - NX[1]);
  nf[pkh(0, 2, pos, nKt, nPt)] = (h16)(X[j * 3 + 2] - NX[2]);
  const float* F = feats + (long)b * Cpc * Np;
  for (int c = 0; c < Cpc; ++c)
    nf[pkh(0, 3 + c, pos, nKt, nPt)] = (h16)(F[(long)c * Np + j] - F[(long)c * Np + cj]);
  for (int c = 3 + Cpc; c < Kpad; ++c)
    nf[pkh(0, c, pos, nKt, nPt)] = (h16)0.f;
}

// max over 32 neighbors on packed layout: -> (B,C,M) f32
__global__ void maxpool_k(const h16* __restrict__ act, float* __restrict__ out,
                          int C, int M, int BM) {
  int t = blockIdx.x * blockDim.x + threadIdx.x;
  if (t >= BATCH * C * M) return;
  int m = t % M, c = (t / M) % C, b = t / (M * C);
  int nKt = C >> 5;
  long nPt = (long)BM * 2;
  long posBase = (long)(b * M + m) * 32;
  float mx = -3.0e38f;
  for (int kk = 0; kk < 32; ++kk)
    mx = fmaxf(mx, (float)act[pkh(0, c, posBase + kk, nKt, nPt)]);
  out[((long)b * C + c) * M + m] = mx;
}

// grouped 1x1 conv; writes planar f32 (for A-side packing) + packed f16 (B side)
__global__ void qk_conv(const float* __restrict__ w, const float* __restrict__ xf,
                        float* __restrict__ qk, h16* __restrict__ qkp, int C, int N) {
  int t = blockIdx.x * blockDim.x + threadIdx.x;
  if (t >= BATCH * C * N) return;
  int n = t % N, oc = (t / N) % C, b = t / (N * C);
  int cg = C >> 3;
  int g = oc / cg, o = oc % cg;
  const float* W  = w + ((long)g * cg + o) * cg;
  const float* Xb = xf + (long)b * C * N + (long)g * cg * N + n;
  float s = 0.f;
  for (int c = 0; c < cg; ++c) s += W[c] * Xb[(long)c * N];
  qk[t] = s;
  qkp[pkh(b, oc, n, C >> 5, (long)(N >> 4))] = (h16)s;
}

__global__ void softmax_rows(float* __restrict__ e, int N) {
  __shared__ float red[256];
  float* p = e + (long)blockIdx.x * N;
  int tid = threadIdx.x;
  float m = -3.0e38f;
  for (int j = tid; j < N; j += blockDim.x) m = fmaxf(m, p[j]);
  red[tid] = m; __syncthreads();
  for (int off = 128; off > 0; off >>= 1) {
    if (tid < off) red[tid] = fmaxf(red[tid], red[tid + off]);
    __syncthreads();
  }
  float mx = red[0]; __syncthreads();
  float s = 0.f;
  for (int j = tid; j < N; j += blockDim.x) { float v = __expf(p[j] - mx); p[j] = v; s += v; }
  red[tid] = s; __syncthreads();
  for (int off = 128; off > 0; off >>= 1) {
    if (tid < off) red[tid] += red[tid + off];
    __syncthreads();
  }
  float inv = 1.f / red[0];
  for (int j = tid; j < N; j += blockDim.x) p[j] *= inv;
}

// attn /= (1e-9 + sum over rows) per column
__global__ void attn_colnorm(float* __restrict__ a, int N) {
  int t = blockIdx.x * blockDim.x + threadIdx.x;
  if (t >= BATCH * N) return;
  int b = t / N, m = t % N;
  float* base = a + (long)b * N * N + m;
  float s = 0.f;
  for (int nn = 0; nn < N; ++nn) s += base[(long)nn * N];
  float inv = 1.f / (1e-9f + s);
  for (int nn = 0; nn < N; ++nn) base[(long)nn * N] *= inv;
}

__global__ void ew_sub(const float* __restrict__ a, const float* __restrict__ b,
                       float* __restrict__ o, long n) {
  long t = (long)blockIdx.x * blockDim.x + threadIdx.x;
  if (t < n) o[t] = a[t] - b[t];
}

// 3-NN with inverse-distance weights
__global__ void three_nn(const float* __restrict__ un, const float* __restrict__ kn,
                         float* __restrict__ w3, int* __restrict__ i3, int n, int mk) {
  int t = blockIdx.x * blockDim.x + threadIdx.x;
  if (t >= BATCH * n) return;
  int b = t / n;
  const float* U = un + (long)t * 3;
  const float* Kp = kn + (long)b * mk * 3;
  float ux = U[0], uy = U[1], uz = U[2];
  float d0 = 3e38f, d1 = 3e38f, d2v = 3e38f;
  int i0 = 0, i1 = 0, i2 = 0;
  for (int mm = 0; mm < mk; ++mm) {
    float dx = Kp[mm * 3 + 0] - ux, dy = Kp[mm * 3 + 1] - uy, dz = Kp[mm * 3 + 2] - uz;
    float d = dx * dx + dy * dy + dz * dz;
    if (d < d0)       { d2v = d1; i2 = i1; d1 = d0; i1 = i0; d0 = d; i0 = mm; }
    else if (d < d1)  { d2v = d1; i2 = i1; d1 = d;  i1 = mm; }
    else if (d < d2v) { d2v = d; i2 = mm; }
  }
  float r0 = 1.f / (fmaxf(d0, 0.f) + 1e-8f);
  float r1 = 1.f / (fmaxf(d1, 0.f) + 1e-8f);
  float r2 = 1.f / (fmaxf(d2v, 0.f) + 1e-8f);
  float inv = 1.f / (r0 + r1 + r2);
  w3[(long)t * 3 + 0] = r0 * inv; w3[(long)t * 3 + 1] = r1 * inv; w3[(long)t * 3 + 2] = r2 * inv;
  i3[(long)t * 3 + 0] = i0; i3[(long)t * 3 + 1] = i1; i3[(long)t * 3 + 2] = i2;
}

// FP input written directly in packed B layout, zero-padded to Kpad channels.
__global__ void build_fp_input(const float* __restrict__ kf, int Ck, int mk,
                               const float* __restrict__ skip, int Cs,
                               const float* __restrict__ w3, const int* __restrict__ i3,
                               h16* __restrict__ cat, int Kpad, int n) {
  long t = (long)blockIdx.x * blockDim.x + threadIdx.x;
  long P = (long)BATCH * n;
  if (t >= (long)Kpad * P) return;
  int c = (int)(t / P);
  long pos = t % P;
  int b = (int)(pos / n);
  int j = (int)(pos % n);
  float val = 0.f;
  if (c < Ck) {
    const float* base = kf + ((long)b * Ck + c) * mk;
    long o = ((long)b * n + j) * 3;
    val = w3[o] * base[i3[o]] + w3[o + 1] * base[i3[o + 1]] + w3[o + 2] * base[i3[o + 2]];
  } else if (c < Ck + Cs) {
    val = skip[((long)b * Cs + (c - Ck)) * n + j];
  }
  cat[pkh(0, c, pos, Kpad >> 5, P >> 4)] = (h16)val;
}

// ---------------------------------------------------------------------------
static inline unsigned cdivl(long a, int b) { return (unsigned)((a + b - 1) / b); }
static inline dim3 ggrid(int O, int P, int Z) {
  return dim3((unsigned)((P + 255) / 256), (unsigned)(O / 16), (unsigned)Z);
}

extern "C" void kernel_launch(void* const* d_in, const int* in_sizes, int n_in,
                              void* d_out, int out_size, void* d_ws, size_t ws_size,
                              hipStream_t stream) {
  const float BN_SCALE = 0.9999950000374997f;  // 1/sqrt(1+1e-5)
  const float* x = (const float*)d_in[0];      // (8,1,4096,3) == (8,4096,3)

  // ---- parameter extraction (setup_inputs dict insertion order) ----
  int cur = 1;
  const float *saW[4][3], *saG[4][3], *saB[4][3];
  const float *qkw[4], *vw[4], *vb[4], *tw[4], *tb[4], *bng[4], *bnb[4];
  for (int i = 0; i < 4; ++i) {
    for (int l = 0; l < 3; ++l) {
      saW[i][l] = (const float*)d_in[cur++];
      saG[i][l] = (const float*)d_in[cur++];
      saB[i][l] = (const float*)d_in[cur++];
    }
    qkw[i] = (const float*)d_in[cur++];
    vw[i]  = (const float*)d_in[cur++];
    vb[i]  = (const float*)d_in[cur++];
    tw[i]  = (const float*)d_in[cur++];
    tb[i]  = (const float*)d_in[cur++];
    bng[i] = (const float*)d_in[cur++];
    bnb[i] = (const float*)d_in[cur++];
  }
  const float *fpW[4][3], *fpG[4][3], *fpB[4][3];
  int fpL[4] = {3, 2, 2, 2};
  for (int j = 0; j < 4; ++j)
    for (int l = 0; l < fpL[j]; ++l) {
      fpW[j][l] = (const float*)d_in[cur++];
      fpG[j][l] = (const float*)d_in[cur++];
      fpB[j][l] = (const float*)d_in[cur++];
    }
  (void)n_in; (void)in_sizes; (void)out_size; (void)ws_size;

  // ---- workspace layout (float units) ----
  float* ws = (float*)d_ws;
  const long OFF_FEAT0 = 0;                                 // 98304
  const long OFF_LF[5] = {0, 98304, 622592, 884736, 1015808};
  const long OFF_XYZ[5] = {0, 1081344, 1105920, 1112064, 1113600};
  const long OFF_CIDX = 1113984;                            // 10880 ints
  const long OFF_KNN  = 1124864;                            // 348160 ints
  const long U0 = 1474560;
  // SA-stage scratch
  h16*   ACT_A  = (h16*)(ws + U0);                          // 16.78M halfs (packed)
  h16*   ACT_B  = (h16*)(ws + U0 + 8388608);                // 16.78M halfs (packed)
  float* ENERGY = ws + U0 + 16777216;                       // 8.39M f32
  float* QK   = ws + U0 + 25165824;                         // 524288
  float* Vb_  = QK + 524288;
  float* XR   = Vb_ + 524288;
  float* DIFF = XR + 524288;
  h16*   PACKA = (h16*)(ws + U0 + 27262976);                // 524288 halfs (packed A)
  h16*   QKP   = (h16*)(ws + U0 + 27525120);                // 524288 halfs (packed qk)
  h16*   PACKX = (h16*)(ws + U0 + 27787264);                // 8.39M halfs (packed X)
  float* NN3W = ws + U0 + 25427968;
  int*   NN3I = (int*)(ws + U0 + 25526272);
  int* CIDX = (int*)(ws + OFF_CIDX);
  int* KNN  = (int*)(ws + OFF_KNN);

  auto PACK = [&](const float* A, long sAo, long sAk, long sAb, int O, int K, int Z) -> long {
    int nOt = O >> 4, nKt = (K + 31) >> 5;
    long tot = (long)Z * nOt * nKt * 32;
    pack_A<<<cdivl(tot, 256), 256, 0, stream>>>(A, sAo, sAk, sAb, PACKA, O, K, Z);
    return (Z > 1) ? (long)nOt * nKt * 32 : 0;
  };
  auto PACKB = [&](const float* X, long sXk, long sXb, int K, int P, int Z) -> long {
    int nKt = (K + 31) >> 5, nPt = P >> 4;
    long tot = (long)Z * nKt * nPt * 32;
    pack_B<<<cdivl(tot, 256), 256, 0, stream>>>(X, sXk, sXb, PACKX, K, P, Z);
    return (Z > 1) ? (long)nKt * nPt * 32 : 0;
  };

  transpose_pc<<<cdivl(BATCH * 4096, 256), 256, 0, stream>>>(x, ws + OFF_FEAT0, 4096);

  // ---- SA stages ----
  int Np[4] = {4096, 1024, 256, 64};
  int Ms[4] = {1024, 256, 64, 16};
  int Cp[4] = {3, 64, 128, 256};
  int H0[4] = {32, 64, 128, 256};
  int CH[4] = {64, 128, 256, 512};
  long cOff[4] = {0, 8192, 10240, 10752};
  long kOff[4] = {0, 262144, 327680, 344064};

  for (int i = 0; i < 4; ++i) {
    const float* xyzPrev  = (i == 0) ? x : ws + OFF_XYZ[i];
    float*       xyzNew   = ws + OFF_XYZ[i + 1];
    const float* featPrev = (i == 0) ? ws + OFF_FEAT0 : ws + OFF_LF[i];
    float*       lf       = ws + OFF_LF[i + 1];
    int* ci = CIDX + cOff[i];
    int* kn = KNN + kOff[i];
    int N = Np[i], M = Ms[i], C = CH[i], h = H0[i];
    int Cin = Cp[i] + 3;
    int Kpad0 = (Cin + 31) & ~31;
    int P = BATCH * M * 32;

    fps_kernel<<<BATCH, 256, 0, stream>>>(xyzPrev, ci, N, M);
    gather_xyz<<<cdivl(BATCH * M, 256), 256, 0, stream>>>(xyzPrev, ci, xyzNew, N, M);
    knn_kernel<<<BATCH * M, 128, 0, stream>>>(xyzPrev, xyzNew, kn, N, M);
    edge_feat<<<cdivl(P, 256), 256, 0, stream>>>(xyzPrev, featPrev, xyzNew, ci, kn,
                                                 ACT_A, N, M, Cp[i], Kpad0, P);
    // shared MLP (P always multiple of 256 -> FULL)
    PACK(saW[i][0], (long)Cin, 1, 0, h, Cin, 1);
    wmma_gemm<h16, true, true, false, true, true, false><<<ggrid(h, P, 1), 128, 0, stream>>>(
        PACKA, 0, ACT_A, 0, ACT_B, 0, 0,
        nullptr, saG[i][0], BN_SCALE, saB[i][0], nullptr, 0, 0, h, Cin, P);
    PACK(saW[i][1], (long)h, 1, 0, h, h, 1);
    wmma_gemm<h16, true, true, false, true, true, false><<<ggrid(h, P, 1), 128, 0, stream>>>(
        PACKA, 0, ACT_B, 0, ACT_A, 0, 0,
        nullptr, saG[i][1], BN_SCALE, saB[i][1], nullptr, 0, 0, h, h, P);
    PACK(saW[i][2], (long)h, 1, 0, C, h, 1);
    wmma_gemm<h16, true, true, false, true, true, false><<<ggrid(C, P, 1), 128, 0, stream>>>(
        PACKA, 0, ACT_A, 0, ACT_B, 0, 0,
        nullptr, saG[i][2], BN_SCALE, saB[i][2], nullptr, 0, 0, C, h, P);
    maxpool_k<<<cdivl((long)BATCH * C * M, 256), 256, 0, stream>>>(ACT_B, lf, C, M, BATCH * M);

    // offset self-attention
    int Na = M;
    bool full = (Na & 63) == 0;
    long stQK = (long)(C >> 5) * (Na >> 4) * 32;
    qk_conv<<<cdivl((long)BATCH * C * Na, 256), 256, 0, stream>>>(qkw[i], lf, QK, QKP, C, Na);
    // energy = qk^T qk
    long stA = PACK(QK, 1, (long)Na, (long)C * Na, Na, C, BATCH);
    if (full)
      wmma_gemm<float, false, true, false, false, false, false><<<ggrid(Na, Na, BATCH), 128, 0, stream>>>(
          PACKA, stA, QKP, stQK, ENERGY, (long)Na, (long)Na * Na,
          nullptr, nullptr, 1.f, nullptr, nullptr, 0, 0, Na, C, Na);
    else
      wmma_gemm<float, false, false, false, false, false, false><<<ggrid(Na, Na, BATCH), 128, 0, stream>>>(
          PACKA, stA, QKP, stQK, ENERGY, (long)Na, (long)Na * Na,
          nullptr, nullptr, 1.f, nullptr, nullptr, 0, 0, Na, C, Na);
    softmax_rows<<<BATCH * Na, 256, 0, stream>>>(ENERGY, Na);
    attn_colnorm<<<cdivl(BATCH * Na, 256), 256, 0, stream>>>(ENERGY, Na);
    // v = v_w x + v_b
    PACK(vw[i], (long)C, 1, 0, C, C, 1);
    long stX = PACKB(lf, (long)Na, (long)C * Na, C, Na, BATCH);
    if (full)
      wmma_gemm<float, false, true, true, false, false, false><<<ggrid(C, Na, BATCH), 128, 0, stream>>>(
          PACKA, 0, PACKX, stX, Vb_, (long)Na, (long)C * Na,
          vb[i], nullptr, 1.f, nullptr, nullptr, 0, 0, C, C, Na);
    else
      wmma_gemm<float, false, false, true, false, false, false><<<ggrid(C, Na, BATCH), 128, 0, stream>>>(
          PACKA, 0, PACKX, stX, Vb_, (long)Na, (long)C * Na,
          vb[i], nullptr, 1.f, nullptr, nullptr, 0, 0, C, C, Na);
    // x_r = v attn
    stA = PACK(Vb_, (long)Na, 1, (long)C * Na, C, Na, BATCH);
    stX = PACKB(ENERGY, (long)Na, (long)Na * Na, Na, Na, BATCH);
    if (full)
      wmma_gemm<float, false, true, false, false, false, false><<<ggrid(C, Na, BATCH), 128, 0, stream>>>(
          PACKA, stA, PACKX, stX, XR, (long)Na, (long)C * Na,
          nullptr, nullptr, 1.f, nullptr, nullptr, 0, 0, C, Na, Na);
    else
      wmma_gemm<float, false, false, false, false, false, false><<<ggrid(C, Na, BATCH), 128, 0, stream>>>(
          PACKA, stA, PACKX, stX, XR, (long)Na, (long)C * Na,
          nullptr, nullptr, 1.f, nullptr, nullptr, 0, 0, C, Na, Na);
    ew_sub<<<cdivl((long)BATCH * C * Na, 256), 256, 0, stream>>>(lf, XR, DIFF, (long)BATCH * C * Na);
    // x = x + relu((t_w diff + t_b) * bn_g*BN_SCALE + bn_b)
    PACK(tw[i], (long)C, 1, 0, C, C, 1);
    stX = PACKB(DIFF, (long)Na, (long)C * Na, C, Na, BATCH);
    if (full)
      wmma_gemm<float, false, true, true, true, true, true><<<ggrid(C, Na, BATCH), 128, 0, stream>>>(
          PACKA, 0, PACKX, stX, lf, (long)Na, (long)C * Na,
          tb[i], bng[i], BN_SCALE, bnb[i], lf, (long)Na, (long)C * Na, C, C, Na);
    else
      wmma_gemm<float, false, false, true, true, true, true><<<ggrid(C, Na, BATCH), 128, 0, stream>>>(
          PACKA, 0, PACKX, stX, lf, (long)Na, (long)C * Na,
          tb[i], bng[i], BN_SCALE, bnb[i], lf, (long)Na, (long)C * Na, C, C, Na);
  }

  // ---- FP stages ----
  float* dout = (float*)d_out;
  const long OUT_OFF[4] = {0, 131072, 655360, 2752512};  // lf3, lf2, lf1, lf0
  int nUn[4] = {64, 256, 1024, 4096};
  int mKn[4] = {16, 64, 256, 1024};
  const float* unX[4]  = {ws + OFF_XYZ[3], ws + OFF_XYZ[2], ws + OFF_XYZ[1], x};
  const float* knX[4]  = {ws + OFF_XYZ[4], ws + OFF_XYZ[3], ws + OFF_XYZ[2], ws + OFF_XYZ[1]};
  const float* kfP[4]  = {ws + OFF_LF[4], dout + 0, dout + 131072, dout + 655360};
  int CkA[4] = {512, 256, 256, 256};
  const float* skipP[4] = {ws + OFF_LF[3], ws + OFF_LF[2], ws + OFF_LF[1], ws + OFF_FEAT0};
  int CsA[4] = {256, 128, 64, 3};
  int fpIdx[4] = {3, 2, 1, 0};

  for (int t4 = 0; t4 < 4; ++t4) {
    int n = nUn[t4], mk = mKn[t4], Ck = CkA[t4], Cs = CsA[t4];
    int Cin = Ck + Cs;
    int Kpad = (Cin + 31) & ~31;
    int P = BATCH * n;
    three_nn<<<cdivl(BATCH * n, 256), 256, 0, stream>>>(unX[t4], knX[t4], NN3W, NN3I, n, mk);
    build_fp_input<<<cdivl((long)Kpad * P, 256), 256, 0, stream>>>(
        kfP[t4], Ck, mk, skipP[t4], Cs, NN3W, NN3I, ACT_A, Kpad, n);
    int jf = fpIdx[t4], L = fpL[jf];
    const h16* Xc = ACT_A;
    int Kd = Cin;
    for (int l = 0; l < L; ++l) {
      bool last = (l == L - 1);
      h16* nxt = (Xc == ACT_A) ? ACT_B : ACT_A;
      PACK(fpW[jf][l], (long)Kd, 1, 0, 256, Kd, 1);
      if (last) {
        wmma_gemm<float, false, true, false, true, true, false><<<ggrid(256, P, 1), 128, 0, stream>>>(
            PACKA, 0, Xc, 0, dout + OUT_OFF[t4], (long)P, 0,
            nullptr, fpG[jf][l], BN_SCALE, fpB[jf][l], nullptr, 0, 0, 256, Kd, P);
      } else {
        wmma_gemm<h16, true, true, false, true, true, false><<<ggrid(256, P, 1), 128, 0, stream>>>(
            PACKA, 0, Xc, 0, nxt, 0, 0,
            nullptr, fpG[jf][l], BN_SCALE, fpB[jf][l], nullptr, 0, 0, 256, Kd, P);
        Xc = nxt;
      }
      Kd = 256;
    }
  }
}
